// MultiheadRelativeAttention_54743653155396
// MI455X (gfx1250) — matrix operand order repaired and checked by reading, not compiled, not run on hardware
//
#include <hip/hip_runtime.h>

// ---------------------------------------------------------------------------
// Multihead relative attention (T5 bias) forward for gfx1250 / MI455X.
// All GEMMs use v_wmma_f32_16x16x32_bf16 (wave32 WMMA), f32 accumulate.
// ---------------------------------------------------------------------------

typedef __bf16 bf16;
typedef __attribute__((ext_vector_type(16))) __bf16 v16bf;
typedef __attribute__((ext_vector_type(8)))  __bf16 v8bf;
typedef __attribute__((ext_vector_type(8)))  float  v8f;

#define LSEQ   2048
#define BATCH  2
#define DMODEL 1024
#define NHEAD  16
#define DK     64
#define DV     64
#define SPAN   128

static __device__ __forceinline__ v8f wmma_bf16(v16bf a, v16bf b, v8f c) {
    return __builtin_amdgcn_wmma_f32_16x16x32_bf16(
        /*neg_a=*/false, a, /*neg_b=*/false, b,
        /*c_mod=*/(short)0, c, /*reuse_a=*/false, /*reuse_b=*/false);
}

// A-fragment: row = lane%16; K chunk [kk+16*half, +16)
static __device__ __forceinline__ v16bf mk_afrag(const bf16* row, int kk, int half) {
    v8bf c0 = *(const v8bf*)(row + kk + 16 * half);
    v8bf c1 = *(const v8bf*)(row + kk + 16 * half + 8);
    return __builtin_shufflevector(c0, c1, 0,1,2,3,4,5,6,7,8,9,10,11,12,13,14,15);
}

// ---------------------------------------------------------------------------
// f32 -> bf16 convert
// ---------------------------------------------------------------------------
__global__ void cvt_f32_bf16(const float* __restrict__ src, bf16* __restrict__ dst, int n) {
    int i = blockIdx.x * blockDim.x + threadIdx.x;
    if (i < n) dst[i] = (bf16)src[i];
}

// ---------------------------------------------------------------------------
// Generic 16x16-tile WMMA GEMM:  C[m,n] = sum_k X[m,k] * W[n,k] + bias[n]
// X: bf16, row m at  b*x_z + m*x_r  (elements);  W: bf16 natural [N][K].
// mode 0: bf16 out at b*o_z + m*o_r + n
// mode 1: bf16 out transposed at b*o_z + n*o_rn + m   (16B vector store)
// mode 2: f32  out at b*o_z + m*o_r + n
// grid.z = BATCH, block = 256 (8 waves), one wave per 16x16 tile.
// ---------------------------------------------------------------------------
__global__ __launch_bounds__(256) void gemm16_bf16(
    const bf16* __restrict__ X, const bf16* __restrict__ W,
    const float* __restrict__ bias, void* __restrict__ outp,
    int M, int N, int K,
    long long x_z, long long x_r, long long o_z, long long o_rn, int mode)
{
    const int b    = blockIdx.z;
    const int wave = threadIdx.x >> 5;
    const int lane = threadIdx.x & 31;
    const int half = lane >> 4;
    const int lr   = lane & 15;

    const int ntn  = N >> 4;
    const int tile = blockIdx.x * 8 + wave;
    if (tile >= (M >> 4) * ntn) return;
    const int tm = tile / ntn;
    const int tn = tile % ntn;

    const bf16* xrow = X + (size_t)b * x_z + (size_t)(tm * 16 + lr) * x_r;
    const bf16* wrow = W + (size_t)(tn * 16 + lr) * K + 16 * half;

    v8f acc = {0.f, 0.f, 0.f, 0.f, 0.f, 0.f, 0.f, 0.f};
    for (int kk = 0; kk < K; kk += 32) {
        v16bf a = mk_afrag(xrow, kk, half);
        v16bf bb = *(const v16bf*)(wrow + kk);   // 32B contiguous per lane
        acc = wmma_bf16(a, bb, acc);
    }

    const int n  = tn * 16 + lr;
    const float bv = bias ? bias[n] : 0.0f;

    if (mode == 0) {
        bf16* o = (bf16*)outp;
        #pragma unroll
        for (int r = 0; r < 8; ++r) {
            int m = tm * 16 + half * 8 + r;
            o[(size_t)b * o_z + (size_t)m * o_rn + n] = (bf16)(acc[r] + bv);
        }
    } else if (mode == 1) {
        bf16* o = (bf16*)outp + (size_t)b * o_z + (size_t)n * o_rn + tm * 16 + half * 8;
        v8bf pk;
        #pragma unroll
        for (int r = 0; r < 8; ++r) pk[r] = (bf16)(acc[r] + bv);
        *(v8bf*)o = pk;                          // contiguous transposed store
    } else {
        float* o = (float*)outp;
        #pragma unroll
        for (int r = 0; r < 8; ++r) {
            int m = tm * 16 + half * 8 + r;
            o[(size_t)b * o_z + (size_t)m * o_rn + n] = acc[r] + bv;
        }
    }
}

// ---------------------------------------------------------------------------
// Fused attention: per (h, b, 16-row l-tile) workgroup (256 threads, 8 waves).
// Phase 1: scores QK^T (WMMA) -> 16x2048 f32 strip in LDS (128 KB).
// Phase 2: + rel bias, softmax over t; write P (f32) to d_out attn region.
// Phase 3: P @ V with WMMA, A-fragments read from LDS, K split across wave
//          pairs, partials combined in LDS; bf16 output for the FC GEMM.
// ---------------------------------------------------------------------------
__global__ __launch_bounds__(256) void attn_fused(
    const bf16* __restrict__ qb, const bf16* __restrict__ kb,
    const bf16* __restrict__ vtb, const float* __restrict__ rel_emb,
    float* __restrict__ attn_out, bf16* __restrict__ Ob)
{
    extern __shared__ float sm[];
    float* S   = sm;                    // [16][2048] score strip
    float* rb  = sm + 16 * LSEQ;        // [272]  rel_emb column for this head
    float* red = rb + 272;              // [256]  softmax reduction scratch
    float* ps  = red + 256;             // [8*256] phase-3 partial C fragments

    const int h  = blockIdx.z / BATCH;
    const int b  = blockIdx.z % BATCH;
    const int l0 = blockIdx.x * 16;
    const int tid  = threadIdx.x;
    const int wave = tid >> 5;
    const int lane = tid & 31;
    const int half = lane >> 4;
    const int lr   = lane & 15;

    for (int i = tid; i < 2 * SPAN + 1; i += 256) rb[i] = rel_emb[i * NHEAD + h];

    // ---- Phase 1: raw scores ----
    const bf16* qrow = qb + ((size_t)(l0 + lr) * BATCH + b) * DMODEL + h * DK;
    v16bf aq0 = mk_afrag(qrow, 0, half);
    v16bf aq1 = mk_afrag(qrow, 32, half);

    for (int it = 0; it < 16; ++it) {
        int t0 = (it * 8 + wave) * 16;
        const bf16* krow = kb + ((size_t)(t0 + lr) * BATCH + b) * DMODEL + h * DK + 16 * half;
        v16bf b0 = *(const v16bf*)(krow);
        v16bf b1 = *(const v16bf*)(krow + 32);
        v8f acc = {0.f, 0.f, 0.f, 0.f, 0.f, 0.f, 0.f, 0.f};
        acc = wmma_bf16(aq0, b0, acc);
        acc = wmma_bf16(aq1, b1, acc);
        #pragma unroll
        for (int r = 0; r < 8; ++r)
            S[(half * 8 + r) * LSEQ + t0 + lr] = acc[r];
    }
    __syncthreads();

    // ---- Phase 2: bias + softmax ----
    const int row = tid >> 4;           // 0..15 (l within tile)
    const int seg = tid & 15;           // column group
    float* Srow = S + row * LSEQ;
    const int l = l0 + row;

    float mx = -3.402823e38f;
    for (int j = 0; j < 128; ++j) {
        int t = seg + 16 * j;
        int d = t - l;
        d = d < -SPAN ? -SPAN : (d > SPAN ? SPAN : d);
        float v = Srow[t] * 0.125f + rb[d + SPAN];
        Srow[t] = v;
        mx = fmaxf(mx, v);
    }
    red[tid] = mx;
    __syncthreads();
    float rmax = red[row * 16];
    for (int j = 1; j < 16; ++j) rmax = fmaxf(rmax, red[row * 16 + j]);
    __syncthreads();

    float ssum = 0.f;
    for (int j = 0; j < 128; ++j) {
        int t = seg + 16 * j;
        float e = __expf(Srow[t] - rmax);
        Srow[t] = e;
        ssum += e;
    }
    red[tid] = ssum;
    __syncthreads();
    float tot = 0.f;
    for (int j = 0; j < 16; ++j) tot += red[row * 16 + j];
    float inv = 1.0f / tot;

    float* arow = attn_out + ((size_t)(h * BATCH + b) * LSEQ + l) * LSEQ;
    for (int j = 0; j < 128; ++j) {
        int t = seg + 16 * j;
        float p = Srow[t] * inv;
        Srow[t] = p;                    // keep normalized P for phase 3
        arow[t] = p;                    // coalesced f32 store of attn output
    }
    __syncthreads();

    // ---- Phase 3: P @ V, K split across wave pairs ----
    const int dvt = wave & 3;           // output dv tile
    const int kh  = wave >> 2;          // K half
    const int dv0 = dvt * 16;
    const bf16* vrow = vtb + ((size_t)(b * NHEAD + h) * DV + dv0 + lr) * LSEQ + 16 * half;

    v8f acc = {0.f, 0.f, 0.f, 0.f, 0.f, 0.f, 0.f, 0.f};
    for (int it = 0; it < 32; ++it) {
        int t0 = kh * 1024 + it * 32;
        const float* sr = S + lr * LSEQ + t0;
        v16bf a;
        #pragma unroll
        for (int j = 0; j < 8; ++j) {
            a[j]     = (bf16)sr[16 * half + j];
            a[8 + j] = (bf16)sr[16 * half + 8 + j];
        }
        v16bf bb = *(const v16bf*)(vrow + t0);
        acc = wmma_bf16(a, bb, acc);
    }
    #pragma unroll
    for (int r = 0; r < 8; ++r) ps[wave * 256 + r * 32 + lane] = acc[r];
    __syncthreads();

    if (wave < 4) {
        #pragma unroll
        for (int r = 0; r < 8; ++r) {
            float v = ps[wave * 256 + r * 32 + lane] + ps[(wave + 4) * 256 + r * 32 + lane];
            int l2 = l0 + half * 8 + r;
            Ob[((size_t)l2 * BATCH + b) * DMODEL + h * DV + dv0 + lr] = (bf16)v;
        }
    }
}

// ---------------------------------------------------------------------------
// Residual + LayerNorm: one block per (l,b) row of 1024.
// ---------------------------------------------------------------------------
__global__ __launch_bounds__(256) void resid_ln(
    const float* __restrict__ y0, const float* __restrict__ query,
    const float* __restrict__ g, const float* __restrict__ beta,
    float* __restrict__ out)
{
    __shared__ float r1[256], r2[256];
    const size_t base = (size_t)blockIdx.x * DMODEL;
    const int tid = threadIdx.x;

    float x[4], s1 = 0.f, s2 = 0.f;
    #pragma unroll
    for (int j = 0; j < 4; ++j) {
        int i = tid + j * 256;
        float v = y0[base + i] + query[base + i];
        x[j] = v; s1 += v; s2 += v * v;
    }
    r1[tid] = s1; r2[tid] = s2;
    __syncthreads();
    for (int off = 128; off > 0; off >>= 1) {
        if (tid < off) { r1[tid] += r1[tid + off]; r2[tid] += r2[tid + off]; }
        __syncthreads();
    }
    float mean = r1[0] * (1.0f / DMODEL);
    float var  = r2[0] * (1.0f / DMODEL) - mean * mean;
    float rstd = rsqrtf(var + 1e-5f);
    #pragma unroll
    for (int j = 0; j < 4; ++j) {
        int i = tid + j * 256;
        out[base + i] = (x[j] - mean) * rstd * g[i] + beta[i];
    }
}

// ---------------------------------------------------------------------------
extern "C" void kernel_launch(void* const* d_in, const int* in_sizes, int n_in,
                              void* d_out, int out_size, void* d_ws, size_t ws_size,
                              hipStream_t stream) {
    (void)in_sizes; (void)n_in; (void)out_size; (void)ws_size;

    const float* query   = (const float*)d_in[0];
    const float* key     = (const float*)d_in[1];
    const float* value   = (const float*)d_in[2];
    const float* wq      = (const float*)d_in[3];
    const float* bq      = (const float*)d_in[4];
    const float* wk      = (const float*)d_in[5];
    const float* bk      = (const float*)d_in[6];
    const float* wv      = (const float*)d_in[7];
    const float* bv      = (const float*)d_in[8];
    const float* fc_w    = (const float*)d_in[9];
    const float* fc_b    = (const float*)d_in[10];
    const float* ln_g    = (const float*)d_in[11];
    const float* ln_b    = (const float*)d_in[12];
    const float* rel_emb = (const float*)d_in[13];

    float* out_y    = (float*)d_out;                               // [L,B,D]
    float* out_attn = out_y + (size_t)LSEQ * BATCH * DMODEL;       // [H,B,L,L]

    const int NX = LSEQ * BATCH * DMODEL;   // 4,194,304 activation elements
    const int NW = DMODEL * DMODEL;         // 1,048,576 weight elements

    // workspace layout (bytes)
    char* ws = (char*)d_ws;
    bf16* xq_b  = (bf16*)(ws);                                 // 8 MB
    bf16* xk_b  = (bf16*)(ws + (size_t)8  * 1024 * 1024);
    bf16* xv_b  = (bf16*)(ws + (size_t)16 * 1024 * 1024);
    bf16* wq_b  = (bf16*)(ws + (size_t)24 * 1024 * 1024);      // 2 MB each
    bf16* wk_b  = (bf16*)(ws + (size_t)26 * 1024 * 1024);
    bf16* wv_b  = (bf16*)(ws + (size_t)28 * 1024 * 1024);
    bf16* fcw_b = (bf16*)(ws + (size_t)30 * 1024 * 1024);
    bf16* q_b   = (bf16*)(ws + (size_t)32 * 1024 * 1024);      // [L,B,H*DK]
    bf16* k_b   = (bf16*)(ws + (size_t)40 * 1024 * 1024);      // [L,B,H*DK]
    bf16* vt_b  = (bf16*)(ws + (size_t)48 * 1024 * 1024);      // [B,H,DV,L]
    bf16* O_b   = (bf16*)(ws + (size_t)56 * 1024 * 1024);      // [L,B,H*DV]
    float* y0   = (float*)(ws + (size_t)64 * 1024 * 1024);     // [L,B,D] f32

    // 1) f32 -> bf16 converts
    cvt_f32_bf16<<<NX / 256, 256, 0, stream>>>(query, xq_b, NX);
    cvt_f32_bf16<<<NX / 256, 256, 0, stream>>>(key,   xk_b, NX);
    cvt_f32_bf16<<<NX / 256, 256, 0, stream>>>(value, xv_b, NX);
    cvt_f32_bf16<<<NW / 256, 256, 0, stream>>>(wq,   wq_b,  NW);
    cvt_f32_bf16<<<NW / 256, 256, 0, stream>>>(wk,   wk_b,  NW);
    cvt_f32_bf16<<<NW / 256, 256, 0, stream>>>(wv,   wv_b,  NW);
    cvt_f32_bf16<<<NW / 256, 256, 0, stream>>>(fc_w, fcw_b, NW);

    // 2) projections: M=2048 (l), N=1024, K=1024, grid.z = b
    dim3 gblk(1024, 1, BATCH);   // (128*64 tiles)/8 waves
    // Q -> natural bf16 [L,B,HK]
    gemm16_bf16<<<gblk, 256, 0, stream>>>(xq_b, wq_b, bq, q_b,
        LSEQ, DMODEL, DMODEL, /*x_z=*/DMODEL, /*x_r=*/BATCH * DMODEL,
        /*o_z=*/DMODEL, /*o_rn=*/BATCH * DMODEL, /*mode=*/0);
    // K -> natural bf16 [L,B,HK]
    gemm16_bf16<<<gblk, 256, 0, stream>>>(xk_b, wk_b, bk, k_b,
        LSEQ, DMODEL, DMODEL, DMODEL, BATCH * DMODEL,
        DMODEL, BATCH * DMODEL, 0);
    // V -> transposed bf16 vt[b,h,dv,L]
    gemm16_bf16<<<gblk, 256, 0, stream>>>(xv_b, wv_b, bv, vt_b,
        LSEQ, DMODEL, DMODEL, DMODEL, BATCH * DMODEL,
        /*o_z=*/(long long)NHEAD * DV * LSEQ, /*o_rn=*/LSEQ, /*mode=*/1);

    // 3) fused attention: grid (L/16, 1, H*B); 140 KB dynamic LDS
    dim3 agrid(LSEQ / 16, 1, NHEAD * BATCH);
    size_t shbytes = (size_t)(16 * LSEQ + 272 + 256 + 8 * 256) * sizeof(float);
    attn_fused<<<agrid, 256, shbytes, stream>>>(q_b, k_b, vt_b, rel_emb,
                                                out_attn, O_b);

    // 4) FC: y0 = O @ fc_w^T + fc_b  (f32 out)
    gemm16_bf16<<<gblk, 256, 0, stream>>>(O_b, fcw_b, fc_b, y0,
        LSEQ, DMODEL, DMODEL, DMODEL, BATCH * DMODEL,
        DMODEL, BATCH * DMODEL, 2);

    // 5) residual + LayerNorm
    resid_ln<<<LSEQ * BATCH, 256, 0, stream>>>(y0, query, ln_g, ln_b, out_y);
}
